// StandardMACE_21139829031607
// MI455X (gfx1250) — matrix operand order repaired and checked by reading, not compiled, not run on hardware
//
#include <hip/hip_runtime.h>
#include <math.h>

// ---------------------------------------------------------------------------
// Problem constants (match reference)
// ---------------------------------------------------------------------------
#define N_ATOMS 8192
#define E_TOT   131072
#define C_CH    64
#define NB_F    8
#define NG_G    16
#define NL_L    2
#define SH_N    16
#define NEL_T   64

typedef __attribute__((ext_vector_type(16))) _Float16 v16h;
typedef __attribute__((ext_vector_type(8)))  _Float16 v8h;
typedef __attribute__((ext_vector_type(8)))  float    v8f;

union AV { v16h v; v8h h8[2]; };

// Fast SiLU: x * sigmoid(x) using hardware v_rcp_f32 (avoids the IEEE
// div_scale/div_fmas/div_fixup expansion).
static __device__ __forceinline__ float silu(float x) {
    return x * __builtin_amdgcn_rcpf(1.0f + __expf(-x));
}

// ---------------------------------------------------------------------------
// 0) init: zero output accumulators and CSR degree array
// ---------------------------------------------------------------------------
__global__ void init_kernel(float* __restrict__ out, int* __restrict__ deg) {
    int i = blockIdx.x * 256 + threadIdx.x;
    if (i < NG_G) out[i] = 0.0f;
    if (i < N_ATOMS) deg[i] = 0;
}

// ---------------------------------------------------------------------------
// 1) geometry: per-edge spherical harmonics Y[16] and radial features feats[8]
// ---------------------------------------------------------------------------
__global__ __launch_bounds__(256) void geom_kernel(
    const float* __restrict__ pos, const int* __restrict__ ei,
    float* __restrict__ Y, float* __restrict__ feats)
{
    int e = blockIdx.x * 256 + threadIdx.x;
    if (e >= E_TOT) return;
    int s = ei[e];
    int d = ei[E_TOT + e];
    float vx = pos[d * 3 + 0] - pos[s * 3 + 0];
    float vy = pos[d * 3 + 1] - pos[s * 3 + 1];
    float vz = pos[d * 3 + 2] - pos[s * 3 + 2];
    float r2 = vx * vx + vy * vy + vz * vz + 1e-12f;
    float ir = __builtin_amdgcn_rsqf(r2);   // v_rsq_f32: 1/sqrt(r2)
    float r  = r2 * ir;                      // r = r2 * rsq(r2)
    float x = vx * ir, y = vy * ir, z = vz * ir;

    const float s3  = 1.7320508075688772f;   // sqrt(3)
    const float s15 = 3.872983346207417f;    // sqrt(15)
    const float s5  = 2.23606797749979f;     // sqrt(5)
    const float s105= 10.246950765959598f;   // sqrt(105)
    const float s7  = 2.6457513110645907f;   // sqrt(7)
    const float c35 = 2.0916500663351889f;   // sqrt(35/8)
    const float c21 = 1.6201851746019651f;   // sqrt(21/8)

    float* Yr = Y + (size_t)e * SH_N;
    Yr[0]  = 1.0f;
    Yr[1]  = s3 * x;
    Yr[2]  = s3 * y;
    Yr[3]  = s3 * z;
    Yr[4]  = s15 * x * y;
    Yr[5]  = s15 * y * z;
    Yr[6]  = 0.5f * s5 * (3.0f * z * z - 1.0f);
    Yr[7]  = s15 * x * z;
    Yr[8]  = 0.5f * s15 * (x * x - y * y);
    Yr[9]  = c35 * y * (3.0f * x * x - y * y);
    Yr[10] = s105 * x * y * z;
    Yr[11] = c21 * y * (5.0f * z * z - 1.0f);
    Yr[12] = 0.5f * s7 * (5.0f * z * z * z - 3.0f * z);
    Yr[13] = c21 * x * (5.0f * z * z - 1.0f);
    Yr[14] = 0.5f * s105 * z * (x * x - y * y);
    Yr[15] = c35 * x * (x * x - 3.0f * y * y);

    // polynomial cutoff, p=6: 1 - 28 x^6 + 48 x^7 - 21 x^8, x = r/5
    float xr = r * 0.2f;
    float cut = 0.0f;
    if (xr < 1.0f) {
        float x2 = xr * xr, x4 = x2 * x2, x6 = x4 * x2;
        cut = 1.0f - 28.0f * x6 + 48.0f * x6 * xr - 21.0f * x6 * x2;
    }
    float pref = 0.6324555320336759f * ir * cut; // sqrt(2/5)/r * cutoff
    const float pi_over_R = 0.6283185307179586f; // pi/5
    float* fr = feats + (size_t)e * NB_F;
#pragma unroll
    for (int k = 1; k <= NB_F; ++k)
        fr[k - 1] = pref * __sinf((float)k * pi_over_R * r);
}

// ---------------------------------------------------------------------------
// 2) embed: h = W_embed[atom_types]
// ---------------------------------------------------------------------------
__global__ void embed_kernel(const int* __restrict__ types,
                             const float* __restrict__ W_embed,
                             float* __restrict__ h)
{
    int idx = blockIdx.x * 256 + threadIdx.x;
    if (idx >= N_ATOMS * C_CH) return;
    int n = idx >> 6, c = idx & 63;
    h[idx] = W_embed[types[n] * C_CH + c];
}

// ---------------------------------------------------------------------------
// 3) CSR build: degree count, exclusive scan, index fill
// ---------------------------------------------------------------------------
__global__ void count_kernel(const int* __restrict__ ei, int* __restrict__ deg) {
    int e = blockIdx.x * 256 + threadIdx.x;
    if (e < E_TOT) atomicAdd(&deg[ei[E_TOT + e]], 1);
}

__global__ __launch_bounds__(1024) void scan_kernel(
    const int* __restrict__ deg, int* __restrict__ rowptr, int* __restrict__ cursor)
{
    __shared__ int part[1024];
    int t = threadIdx.x;
    int loc[8];
    int s = 0;
#pragma unroll
    for (int j = 0; j < 8; ++j) { loc[j] = s; s += deg[t * 8 + j]; }
    part[t] = s;
    __syncthreads();
    for (int off = 1; off < 1024; off <<= 1) {
        int add = (t >= off) ? part[t - off] : 0;
        int v = part[t];
        __syncthreads();
        part[t] = v + add;
        __syncthreads();
    }
    int base = (t == 0) ? 0 : part[t - 1];
#pragma unroll
    for (int j = 0; j < 8; ++j) {
        int v = base + loc[j];
        rowptr[t * 8 + j] = v;
        cursor[t * 8 + j] = v;
    }
    if (t == 1023) rowptr[N_ATOMS] = part[1023];
}

__global__ void fill_kernel(const int* __restrict__ ei, int* __restrict__ cursor,
                            int* __restrict__ edge_of)
{
    int e = blockIdx.x * 256 + threadIdx.x;
    if (e >= E_TOT) return;
    int d = ei[E_TOT + e];
    int p = atomicAdd(&cursor[d], 1);
    edge_of[p] = e;
}

// ---------------------------------------------------------------------------
// 4) edge MLP (WMMA): Rw = (silu(silu(feats@W1)@W2))@W3 ; rwh = Rw * h[src]
//    Per block: 128 edges. Per wave: 16 edges x 64 channels.
//    All 8 WMMAs of a stage are issued back-to-back (independent
//    accumulators), then the activation pass runs -- hides the
//    WMMA->VALU hazard window and keeps the matrix pipe busy.
// ---------------------------------------------------------------------------
#define EPB 128

__global__ __launch_bounds__(256) void edge_mlp_kernel(
    const float* __restrict__ feats, const float* __restrict__ h,
    const int* __restrict__ srcIdx,
    const float* __restrict__ W1, const float* __restrict__ W2,
    const float* __restrict__ W3, float* __restrict__ rwh)
{
    __shared__ __align__(16) _Float16 w2t[C_CH * C_CH];   // N-major W2^T
    __shared__ __align__(16) _Float16 w3t[C_CH * C_CH];   // N-major W3^T
    __shared__ float w1s[NB_F * C_CH];
    __shared__ float ftile[EPB * NB_F];
    __shared__ __align__(16) _Float16 act1[EPB * C_CH];
    __shared__ __align__(16) _Float16 act2[8][16 * C_CH];
    __shared__ int sSrc[EPB];

    const int t  = threadIdx.x;
    const int e0 = blockIdx.x * EPB;

    // ---- cooperative loads: weights -> LDS (f16, transposed to N-major) ----
    for (int idx = t; idx < C_CH * C_CH; idx += 256) {
        int k = idx >> 6, n = idx & 63;
        w2t[n * C_CH + k] = (_Float16)W2[idx];
        w3t[n * C_CH + k] = (_Float16)W3[idx];
    }
    for (int idx = t; idx < NB_F * C_CH; idx += 256) w1s[idx] = W1[idx];
    for (int idx = t; idx < EPB * NB_F; idx += 256)  ftile[idx] = feats[(size_t)e0 * NB_F + idx];
    if (t < EPB) sSrc[t] = srcIdx[e0 + t];
    __syncthreads();

    // ---- stage 0: X1 = silu(feats @ W1), K=8 (VALU, negligible FLOPs) ----
    for (int idx = t; idx < EPB * C_CH; idx += 256) {
        int el = idx >> 6, n = idx & 63;
        float acc = 0.0f;
#pragma unroll
        for (int k = 0; k < NB_F; ++k) acc += ftile[el * NB_F + k] * w1s[k * C_CH + n];
        act1[idx] = (_Float16)silu(acc);
    }
    __syncthreads();

    const int wave   = t >> 5;
    const int lane   = t & 31;
    const int lrow   = lane & 15;   // M (for A) / N (for B,C)
    const int laneHi = lane >> 4;
    const int mBase  = wave * 16;

    // A-matrix 16x32 f16 layout: lane holds row M=lrow; element e -> K =
    // kt*32 + (e<8?0:16) + laneHi*8 + (e&7) => two contiguous 16B chunks.
    auto loadA = [&](const _Float16* row, int kt) -> v16h {
        AV av;
        av.h8[0] = *(const v8h*)(row + kt * 32 + laneHi * 8);
        av.h8[1] = *(const v8h*)(row + kt * 32 + 16 + laneHi * 8);
        return av.v;
    };
    // B-matrix 32x16 f16 layout: lane holds col N=lrow; element e -> K =
    // kt*32 + (e<8?0:16) + laneHi*8 + (e&7) => two 16B chunks of the N-major weight row.
    auto loadB = [&](const _Float16* wts, int ncol, int kt) -> v16h {
        AV av;
        const _Float16* p = wts + ncol * C_CH + kt * 32 + laneHi * 8;
        av.h8[0] = *(const v8h*)(p);
        av.h8[1] = *(const v8h*)(p + 16);
        return av.v;
    };

    v8f acc[4];

    // ---- stage 1: X2 = silu(X1 @ W2) via WMMA burst ----
    {
        const _Float16* aRow = act1 + (mBase + lrow) * C_CH;
        v16h a0 = loadA(aRow, 0);
        v16h a1 = loadA(aRow, 1);
#pragma unroll
        for (int nt = 0; nt < 4; ++nt) {
            v16h b0 = loadB(w2t, nt * 16 + lrow, 0);
            v16h b1 = loadB(w2t, nt * 16 + lrow, 1);
            v8f c = {};
            c = __builtin_amdgcn_wmma_f32_16x16x32_f16(false, a0, false, b0, (short)0, c, false, false);
            c = __builtin_amdgcn_wmma_f32_16x16x32_f16(false, a1, false, b1, (short)0, c, false, false);
            acc[nt] = c;
        }
        // batched SiLU: expose 32 independent exp->add->rcp chains
        float sg[4][8];
#pragma unroll
        for (int nt = 0; nt < 4; ++nt)
#pragma unroll
            for (int r = 0; r < 8; ++r) sg[nt][r] = __expf(-acc[nt][r]);
#pragma unroll
        for (int nt = 0; nt < 4; ++nt)
#pragma unroll
            for (int r = 0; r < 8; ++r) sg[nt][r] = __builtin_amdgcn_rcpf(1.0f + sg[nt][r]);
#pragma unroll
        for (int nt = 0; nt < 4; ++nt)
#pragma unroll
            for (int r = 0; r < 8; ++r)
                act2[wave][(r + laneHi * 8) * C_CH + nt * 16 + lrow] =
                    (_Float16)(acc[nt][r] * sg[nt][r]);
    }
    __syncthreads();

    // ---- stage 2: Rw = X2 @ W3 via WMMA burst; fuse rwh = Rw * h[src] ----
    {
        const _Float16* aRow = act2[wave] + lrow * C_CH;
        v16h a0 = loadA(aRow, 0);
        v16h a1 = loadA(aRow, 1);
#pragma unroll
        for (int nt = 0; nt < 4; ++nt) {
            v16h b0 = loadB(w3t, nt * 16 + lrow, 0);
            v16h b1 = loadB(w3t, nt * 16 + lrow, 1);
            v8f c = {};
            c = __builtin_amdgcn_wmma_f32_16x16x32_f16(false, a0, false, b0, (short)0, c, false, false);
            c = __builtin_amdgcn_wmma_f32_16x16x32_f16(false, a1, false, b1, (short)0, c, false, false);
            acc[nt] = c;
        }
#pragma unroll
        for (int nt = 0; nt < 4; ++nt) {
#pragma unroll
            for (int r = 0; r < 8; ++r) {
                int eLoc = mBase + r + laneHi * 8;
                int ch   = nt * 16 + lrow;
                int s    = sSrc[eLoc];
                float hv = h[(size_t)s * C_CH + ch];
                rwh[(size_t)(e0 + eLoc) * C_CH + ch] = acc[nt][r] * hv;
            }
        }
    }
}

// ---------------------------------------------------------------------------
// 5) node gather + update (fused): A = CSR-gather(rwh ⊗ Y)/16; invariants;
//    h = inv·w_prod[type] + h @ W_sc[type]; readout energy -> atomic out.
//    One block (256 threads) per node. Thread t owns (c = t/4, 4 SH slots).
// ---------------------------------------------------------------------------
__global__ __launch_bounds__(256) void node_kernel(
    const float* __restrict__ rwh, const float* __restrict__ Y,
    const int* __restrict__ edge_of, const int* __restrict__ rowptr,
    float* __restrict__ h, const float* __restrict__ Wsc,
    const float* __restrict__ wprod, const int* __restrict__ types,
    const int* __restrict__ batch,
    const float* __restrict__ w_ro0, const float* __restrict__ W_ro1a,
    const float* __restrict__ w_ro1b, int layer, float* __restrict__ out)
{
    __shared__ float srwh[8 * C_CH];
    __shared__ float sY[8 * SH_N];
    __shared__ int   sE[8];
    __shared__ float As[C_CH * SH_N];
    __shared__ float hs[C_CH];
    __shared__ float hn[C_CH];
    __shared__ float red[C_CH];

    const int n = blockIdx.x;
    const int t = threadIdx.x;
    const int c = t >> 2, q = t & 3;

    const int ty = types[n];
    const float* Wm = Wsc + (size_t)ty * C_CH * C_CH;
    // Pull this node's 16KB type matrix toward the WGP while we gather
    // (global_prefetch_b8; 64 threads cover all 64 cache lines).
    if (t < C_CH) __builtin_prefetch(Wm + t * C_CH, 0, 0);

    if (t < C_CH) hs[t] = h[(size_t)n * C_CH + t];

    float a0 = 0.f, a1 = 0.f, a2 = 0.f, a3 = 0.f;
    const int beg = rowptr[n], end = rowptr[n + 1];
    for (int base = beg; base < end; base += 8) {
        int cnt = min(8, end - base);
        __syncthreads();
        if (t < cnt) sE[t] = edge_of[base + t];
        __syncthreads();
        for (int off = t; off < cnt * C_CH; off += 256) {
            int le = off >> 6, cc = off & 63;
            srwh[off] = rwh[(size_t)sE[le] * C_CH + cc];
        }
        for (int off = t; off < cnt * SH_N; off += 256) {
            int le = off >> 4, ss = off & 15;
            sY[off] = Y[(size_t)sE[le] * SH_N + ss];
        }
        __syncthreads();
        for (int le = 0; le < cnt; ++le) {
            float m = srwh[le * C_CH + c];
            a0 += m * sY[le * SH_N + q * 4 + 0];
            a1 += m * sY[le * SH_N + q * 4 + 1];
            a2 += m * sY[le * SH_N + q * 4 + 2];
            a3 += m * sY[le * SH_N + q * 4 + 3];
        }
    }
    const float inv_avg = 1.0f / 16.0f; // 1/AVG_NEI
    As[c * SH_N + q * 4 + 0] = a0 * inv_avg;
    As[c * SH_N + q * 4 + 1] = a1 * inv_avg;
    As[c * SH_N + q * 4 + 2] = a2 * inv_avg;
    As[c * SH_N + q * 4 + 3] = a3 * inv_avg;
    __syncthreads();

    if (t < C_CH) {
        const int d = t;
        float i0 = As[d * SH_N + 0];
        float i1 = 0.f, i2 = 0.f, i3 = 0.f;
#pragma unroll
        for (int s = 1; s < 4; ++s)  { float v = As[d * SH_N + s]; i1 += v * v; }
#pragma unroll
        for (int s = 4; s < 9; ++s)  { float v = As[d * SH_N + s]; i2 += v * v; }
#pragma unroll
        for (int s = 9; s < 16; ++s) { float v = As[d * SH_N + s]; i3 += v * v; }
        const float* wp = wprod + ty * 4;
        float val = i0 * wp[0] + i1 * wp[1] + i2 * wp[2] + i3 * wp[3];
        float sc = 0.f;
        for (int k = 0; k < C_CH; ++k) sc += hs[k] * Wm[k * C_CH + d];
        val += sc;
        hn[d] = val;
        h[(size_t)n * C_CH + d] = val;
    }
    __syncthreads();

    if (layer == 0) {
        if (t < C_CH) red[t] = hn[t] * w_ro0[t];
        __syncthreads();
        if (t == 0) {
            float s = 0.f;
            for (int k = 0; k < C_CH; ++k) s += red[k];
            atomicAdd(&out[batch[n]], s);
        }
    } else {
        if (t < 16) {
            float z = 0.f;
            for (int k = 0; k < C_CH; ++k) z += hn[k] * W_ro1a[k * 16 + t];
            red[t] = silu(z) * w_ro1b[t];
        }
        __syncthreads();
        if (t == 0) {
            float s = 0.f;
            for (int k = 0; k < 16; ++k) s += red[k];
            atomicAdd(&out[batch[n]], s);
        }
    }
}

// ---------------------------------------------------------------------------
// Host-side launch
// ---------------------------------------------------------------------------
extern "C" void kernel_launch(void* const* d_in, const int* in_sizes, int n_in,
                              void* d_out, int out_size, void* d_ws, size_t ws_size,
                              hipStream_t stream)
{
    (void)in_sizes; (void)n_in; (void)out_size; (void)ws_size;
    const float* pos     = (const float*)d_in[0];
    const int*   types   = (const int*)d_in[1];
    const int*   ei      = (const int*)d_in[2];   // (2,E)
    const int*   batch   = (const int*)d_in[3];
    const float* W_embed = (const float*)d_in[4];
    const float* W_r1    = (const float*)d_in[5]; // (2,8,64)
    const float* W_r2    = (const float*)d_in[6]; // (2,64,64)
    const float* W_r3    = (const float*)d_in[7]; // (2,64,64)
    const float* W_sc    = (const float*)d_in[8]; // (2,64,64,64)
    const float* w_prod  = (const float*)d_in[9]; // (2,64,4)
    const float* w_ro0   = (const float*)d_in[10];
    const float* W_ro1a  = (const float*)d_in[11];
    const float* w_ro1b  = (const float*)d_in[12];
    float* out = (float*)d_out;

    // workspace carve-up (~48 MB)
    float* Yb      = (float*)d_ws;                 // E*16
    float* feats   = Yb + (size_t)E_TOT * SH_N;    // E*8
    float* hbuf    = feats + (size_t)E_TOT * NB_F; // N*64
    float* rwh     = hbuf + (size_t)N_ATOMS * C_CH;// E*64
    int*   deg     = (int*)(rwh + (size_t)E_TOT * C_CH);
    int*   rowptr  = deg + N_ATOMS;                // N+1
    int*   cursor  = rowptr + (N_ATOMS + 1);
    int*   edge_of = cursor + N_ATOMS;             // E

    init_kernel<<<(N_ATOMS + 255) / 256, 256, 0, stream>>>(out, deg);
    geom_kernel<<<E_TOT / 256, 256, 0, stream>>>(pos, ei, Yb, feats);
    embed_kernel<<<(N_ATOMS * C_CH) / 256, 256, 0, stream>>>(types, W_embed, hbuf);
    count_kernel<<<E_TOT / 256, 256, 0, stream>>>(ei, deg);
    scan_kernel<<<1, 1024, 0, stream>>>(deg, rowptr, cursor);
    fill_kernel<<<E_TOT / 256, 256, 0, stream>>>(ei, cursor, edge_of);

    for (int layer = 0; layer < NL_L; ++layer) {
        edge_mlp_kernel<<<E_TOT / EPB, 256, 0, stream>>>(
            feats, hbuf, ei,
            W_r1 + (size_t)layer * NB_F * C_CH,
            W_r2 + (size_t)layer * C_CH * C_CH,
            W_r3 + (size_t)layer * C_CH * C_CH,
            rwh);
        node_kernel<<<N_ATOMS, 256, 0, stream>>>(
            rwh, Yb, edge_of, rowptr, hbuf,
            W_sc + (size_t)layer * NEL_T * C_CH * C_CH,
            w_prod + (size_t)layer * NEL_T * 4,
            types, batch, w_ro0, W_ro1a, w_ro1b, layer, out);
    }
}
